// MeshGraphEncoder_59820304499050
// MI455X (gfx1250) — matrix operand order repaired and checked by reading, not compiled, not run on hardware
//
#include <hip/hip_runtime.h>

#define D      128
#define NT     8        // N tiles (D/16)
#define WAVES  8
#define BLOCK  256
#define TILE_M 128      // rows per block (16 per wave)

typedef __attribute__((ext_vector_type(16))) __bf16 v16bf;
typedef __attribute__((ext_vector_type(8)))  __bf16 v8bf;
typedef __attribute__((ext_vector_type(4)))  int    v4i;
typedef __attribute__((ext_vector_type(8)))  float  v8f;

__device__ __forceinline__ v8f wmma_bf16(v16bf a, v16bf b, v8f c) {
  return __builtin_amdgcn_wmma_f32_16x16x32_bf16(false, a, false, b, (short)0, c,
                                                 false, false);
}

// Build an A fragment from a global f32 row.  Per ISA 16-bit A layout:
// lanes 0-15  : elems 0-7 -> K+0..7,  elems 8-15 -> K+16..23
// lanes 16-31 : elems 0-7 -> K+8..15, elems 8-15 -> K+24..31
// caller passes p already offset by (lane<16 ? 0 : 8); chunks at +0 and +16.
__device__ __forceinline__ v16bf load_a_f32(const float* p) {
  const float4* q = (const float4*)p;
  float4 x0 = q[0], x1 = q[1], x2 = q[4], x3 = q[5];
  v16bf a;
  a[0]=(__bf16)x0.x;  a[1]=(__bf16)x0.y;  a[2]=(__bf16)x0.z;  a[3]=(__bf16)x0.w;
  a[4]=(__bf16)x1.x;  a[5]=(__bf16)x1.y;  a[6]=(__bf16)x1.z;  a[7]=(__bf16)x1.w;
  a[8]=(__bf16)x2.x;  a[9]=(__bf16)x2.y;  a[10]=(__bf16)x2.z; a[11]=(__bf16)x2.w;
  a[12]=(__bf16)x3.x; a[13]=(__bf16)x3.y; a[14]=(__bf16)x3.z; a[15]=(__bf16)x3.w;
  return a;
}

// One-shot: convert W [K x D] (row-major f32) into bf16 WMMA B-fragment order.
// B layout (32x16, 16-bit): lanes 0-15 hold K 0..7,16..23 of column N=lane,
// lanes 16-31 hold K 8..15,24..31 of column N=lane-16 (matches A table order).
__global__ void conv_wfrag(const float* __restrict__ W, __bf16* __restrict__ dst,
                           int ksteps) {
  int idx = blockIdx.x * blockDim.x + threadIdx.x;
  int total = ksteps * NT * 512;
  if (idx >= total) return;
  int e    = idx & 15;
  int lane = (idx >> 4) & 31;
  int nt   = (idx >> 9) & 7;
  int ks   = idx >> 12;
  int k    = ks * 32 + ((e < 8) ? e : e + 8) + ((lane < 16) ? 0 : 8);
  int col  = nt * 16 + (lane & 15);
  dst[idx] = (__bf16)W[k * D + col];
}

// Linear bf16 fragment copy  global(ws) -> LDS, 16B per thread per iteration.
__device__ __forceinline__ void copy_to_lds(const __bf16* __restrict__ g,
                                            __bf16* l, int elems, int tid) {
#if __has_builtin(__builtin_amdgcn_global_load_async_to_lds_b128)
  for (int i = tid * 8; i < elems; i += BLOCK * 8) {
    __builtin_amdgcn_global_load_async_to_lds_b128(
        (__attribute__((address_space(1))) v4i*)(g + i),
        (__attribute__((address_space(3))) v4i*)(l + i), 0, 0);
  }
#else
  for (int i = tid * 8; i < elems; i += BLOCK * 8) {
    *(v8bf*)(l + i) = *(const v8bf*)(g + i);
  }
#endif
}

__device__ __forceinline__ void wait_async_copies() {
#if __has_builtin(__builtin_amdgcn_global_load_async_to_lds_b128)
#if __has_builtin(__builtin_amdgcn_s_wait_asynccnt)
  __builtin_amdgcn_s_wait_asynccnt(0);
#elif defined(__gfx1250__)
  asm volatile("s_wait_asynccnt 0" ::: "memory");
#endif
#endif
}

// MODE 0: edge  (A = [efeat | grid[src] | mesh[dst]], atomic scatter to out=agg)
// MODE 1: mesh  (A = [agg | mesh], residual add, direct store)
// MODE 2: grid  (A = grid, residual add, direct store)
template<int KSTEPS, int MODE>
__global__ __launch_bounds__(BLOCK) void fused_mlp_ln(
    const float* __restrict__ in0, const float* __restrict__ in1,
    const float* __restrict__ in2,
    const int* __restrict__ sidx, const int* __restrict__ didx,
    const __bf16* __restrict__ wf0,   // W0 fragments (bf16, frag-linear)
    const __bf16* __restrict__ wf1,   // W1 fragments
    const float* __restrict__ b0, const float* __restrict__ b1,
    const float* __restrict__ gam, const float* __restrict__ bet,
    const float* __restrict__ resid, float* __restrict__ out, int nrows)
{
  __shared__ alignas(32) __bf16 lds_w0[KSTEPS * NT * 512];
  __shared__ alignas(32) __bf16 lds_w1[4 * NT * 512];
  __shared__ alignas(32) __bf16 h_lds[WAVES][16][D];

  const int tid  = threadIdx.x;
  const int lane = tid & 31;
  const int w    = tid >> 5;
  const int l15  = lane & 15;
  const bool lo  = lane < 16;
  const int hi8  = lo ? 0 : 8;

  copy_to_lds(wf0, lds_w0, KSTEPS * NT * 512, tid);
  copy_to_lds(wf1, lds_w1, 4 * NT * 512, tid);

  const int tile = blockIdx.x * TILE_M;
  int rowA = tile + w * 16 + l15;              // row this lane feeds into A frags
  if (rowA > nrows - 1) rowA = nrows - 1;      // clamp for ragged last tile
  const size_t rb = (size_t)rowA * D;

  const float* p0 = in0 + rb;
  const float* p1 = nullptr;
  const float* p2 = nullptr;
  if (MODE == 0) {
    p1 = in1 + (size_t)sidx[rowA] * D;
    p2 = in2 + (size_t)didx[rowA] * D;
    __builtin_prefetch(p1, 0, 0);              // hide gather latency behind
    __builtin_prefetch(p2, 0, 0);              // the weight->LDS fill
  } else if (MODE == 1) {
    p1 = in1 + rb;
  }

  wait_async_copies();
  __syncthreads();   // weight fragments visible to all waves

  // ---------------- GEMM 1: cat @ W0 ----------------
  v8f acc[NT];
  #pragma unroll
  for (int n = 0; n < NT; ++n) { v8f z = {}; acc[n] = z; }

  #pragma unroll
  for (int ks = 0; ks < KSTEPS; ++ks) {
    const float* p;
    if (MODE == 0)      p = (ks < 4) ? p0 + ks * 32
                          : ((ks < 8) ? p1 + (ks - 4) * 32 : p2 + (ks - 8) * 32);
    else if (MODE == 1) p = (ks < 4) ? p0 + ks * 32 : p1 + (ks - 4) * 32;
    else                p = p0 + ks * 32;
    v16bf a = load_a_f32(p + hi8);
    #pragma unroll
    for (int n = 0; n < NT; ++n) {
      v16bf b = *(const v16bf*)&lds_w0[(ks * NT + n) * 512 + lane * 16];
      acc[n] = wmma_bf16(a, b, acc[n]);
    }
  }

  // bias + SiLU -> bf16 hidden tile in wave-private LDS (C layout -> row major)
  #pragma unroll
  for (int n = 0; n < NT; ++n) {
    float bb = b0[n * 16 + l15];
    #pragma unroll
    for (int j = 0; j < 8; ++j) {
      float v = acc[n][j] + bb;
      float s = v * __fdividef(1.0f, 1.0f + __expf(-v));   // SiLU (fast rcp)
      h_lds[w][lo ? j : j + 8][n * 16 + l15] = (__bf16)s;
    }
  }
  __syncthreads();

  // ---------------- GEMM 2: h @ W1 ----------------
  v8f acc2[NT];
  #pragma unroll
  for (int n = 0; n < NT; ++n) { v8f z = {}; acc2[n] = z; }

  #pragma unroll
  for (int ks = 0; ks < 4; ++ks) {
    const __bf16* hp = &h_lds[w][l15][ks * 32 + hi8];
    v8bf alo = *(const v8bf*)hp;
    v8bf ahi = *(const v8bf*)(hp + 16);
    v16bf a;
    #pragma unroll
    for (int i = 0; i < 8; ++i) { a[i] = alo[i]; a[i + 8] = ahi[i]; }
    #pragma unroll
    for (int n = 0; n < NT; ++n) {
      v16bf b = *(const v16bf*)&lds_w1[(ks * NT + n) * 512 + lane * 16];
      acc2[n] = wmma_bf16(a, b, acc2[n]);
    }
  }

  // ---------------- bias + LayerNorm ----------------
  float sum[8], sq[8];
  #pragma unroll
  for (int j = 0; j < 8; ++j) { sum[j] = 0.f; sq[j] = 0.f; }
  #pragma unroll
  for (int n = 0; n < NT; ++n) {
    float bb = b1[n * 16 + l15];
    #pragma unroll
    for (int j = 0; j < 8; ++j) {
      float v = acc2[n][j] + bb;
      acc2[n][j] = v;
      sum[j] += v; sq[j] += v * v;
    }
  }
  // per-row reduction: rows live in 16-lane half-groups -> xor masks 1,2,4,8
  #pragma unroll
  for (int j = 0; j < 8; ++j) {
    #pragma unroll
    for (int m = 1; m < 16; m <<= 1) {
      sum[j] += __shfl_xor(sum[j], m, 32);
      sq[j]  += __shfl_xor(sq[j],  m, 32);
    }
  }

  float gamc[NT], betc[NT];
  #pragma unroll
  for (int n = 0; n < NT; ++n) {
    gamc[n] = gam[n * 16 + l15];
    betc[n] = bet[n * 16 + l15];
  }

  #pragma unroll
  for (int j = 0; j < 8; ++j) {
    float mean = sum[j] * 0.0078125f;                       // /128
    float rs   = rsqrtf(sq[j] * 0.0078125f - mean * mean + 1e-5f);
    int r = tile + w * 16 + (lo ? j : j + 8);
    if (MODE == 0) {
      int dn = didx[r];                                     // E % 128 == 0
      #pragma unroll
      for (int n = 0; n < NT; ++n) {
        float o = (acc2[n][j] - mean) * rs * gamc[n] + betc[n];
        unsafeAtomicAdd(&out[(size_t)dn * D + n * 16 + l15], o);
      }
    } else if (r < nrows) {
      #pragma unroll
      for (int n = 0; n < NT; ++n) {
        int col = n * 16 + l15;
        float o = (acc2[n][j] - mean) * rs * gamc[n] + betc[n];
        out[(size_t)r * D + col] = resid[(size_t)r * D + col] + o;
      }
    }
  }
}

__global__ void zero_f32(float* p, int n) {
  int i = blockIdx.x * blockDim.x + threadIdx.x;
  if (i < n) p[i] = 0.f;
}

extern "C" void kernel_launch(void* const* d_in, const int* in_sizes, int n_in,
                              void* d_out, int out_size, void* d_ws, size_t ws_size,
                              hipStream_t stream) {
  const float* efeat = (const float*)d_in[0];
  const float* gridf = (const float*)d_in[1];
  const float* meshf = (const float*)d_in[2];
  const int*   sidx  = (const int*)d_in[3];
  const int*   didx  = (const int*)d_in[4];
  const float* eW0 = (const float*)d_in[5];  const float* eb0 = (const float*)d_in[6];
  const float* eW1 = (const float*)d_in[7];  const float* eb1 = (const float*)d_in[8];
  const float* eg  = (const float*)d_in[9];  const float* ebt = (const float*)d_in[10];
  const float* sW0 = (const float*)d_in[11]; const float* sb0 = (const float*)d_in[12];
  const float* sW1 = (const float*)d_in[13]; const float* sb1 = (const float*)d_in[14];
  const float* sg  = (const float*)d_in[15]; const float* sbt = (const float*)d_in[16];
  const float* dW0 = (const float*)d_in[17]; const float* db0 = (const float*)d_in[18];
  const float* dW1 = (const float*)d_in[19]; const float* db1 = (const float*)d_in[20];
  const float* dg  = (const float*)d_in[21]; const float* dbt = (const float*)d_in[22];

  const int NG = 131072, NM = 40962, E = 524288;

  float*  agg      = (float*)d_ws;              // [NM, D] scatter accumulator
  __bf16* frags    = (__bf16*)(agg + (size_t)NM * D);
  __bf16* f_eW0 = frags;                        // 12*NT*512 = 49152
  __bf16* f_eW1 = f_eW0 + 49152;                // 16384
  __bf16* f_sW0 = f_eW1 + 16384;                // 16384
  __bf16* f_sW1 = f_sW0 + 16384;                // 16384
  __bf16* f_dW0 = f_sW1 + 16384;                // 32768
  __bf16* f_dW1 = f_dW0 + 32768;                // 16384

  float* grid_out = (float*)d_out;              // [NG, D]
  float* mesh_out = grid_out + (size_t)NG * D;  // [NM, D]

  int aggN = NM * D;
  zero_f32<<<(aggN + 255) / 256, 256, 0, stream>>>(agg, aggN);

  // one-shot weight -> bf16 fragment conversion
  conv_wfrag<<<(49152 + 255) / 256, 256, 0, stream>>>(eW0, f_eW0, 12);
  conv_wfrag<<<(16384 + 255) / 256, 256, 0, stream>>>(eW1, f_eW1, 4);
  conv_wfrag<<<(16384 + 255) / 256, 256, 0, stream>>>(sW0, f_sW0, 4);
  conv_wfrag<<<(16384 + 255) / 256, 256, 0, stream>>>(sW1, f_sW1, 4);
  conv_wfrag<<<(32768 + 255) / 256, 256, 0, stream>>>(dW0, f_dW0, 8);
  conv_wfrag<<<(16384 + 255) / 256, 256, 0, stream>>>(dW1, f_dW1, 4);

  // edge MLP + scatter-sum into agg (Kin = 3D -> 12 K-steps)
  fused_mlp_ln<12, 0><<<E / TILE_M, BLOCK, 0, stream>>>(
      efeat, gridf, meshf, sidx, didx,
      f_eW0, f_eW1, eb0, eb1, eg, ebt, nullptr, agg, E);

  // grid node MLP + residual (Kin = D -> 4 K-steps)
  fused_mlp_ln<4, 2><<<NG / TILE_M, BLOCK, 0, stream>>>(
      gridf, nullptr, nullptr, nullptr, nullptr,
      f_sW0, f_sW1, sb0, sb1, sg, sbt, gridf, grid_out, NG);

  // mesh node MLP on [agg | mesh] + residual (Kin = 2D -> 8 K-steps)
  fused_mlp_ln<8, 1><<<(NM + TILE_M - 1) / TILE_M, BLOCK, 0, stream>>>(
      agg, meshf, nullptr, nullptr, nullptr,
      f_dW0, f_dW1, db0, db1, dg, dbt, meshf, mesh_out, NM);

  (void)in_sizes; (void)n_in; (void)out_size; (void)ws_size;
}
